// QuadraticModel_16587163697190
// MI455X (gfx1250) — matrix operand order repaired and checked by reading, not compiled, not run on hardware
//
#include <hip/hip_runtime.h>

// ---------------------------------------------------------------------------
// QuadraticModel on MI455X (gfx1250, wave32):
//   h   = quad_expand(x) @ W0.T + b0 ; out = quad_expand(h) @ W1.T + b1
// Weights (2 x 270 MB fp32) dominate -> stream once via async-to-LDS (B128,
// ASYNCcnt), software-pipelined against v_wmma_f32_16x16x32_bf16 (fp32 acc).
// Split-K (8 segments) raises the grid to 1024 blocks / 4096 waves so the
// HBM stream has enough outstanding tiles; partials land via native
// global_atomic_add_f32, bias pre-initialized into the output.
// ---------------------------------------------------------------------------

#define DIM        512
#define NTRI       ((DIM * DIM + DIM) / 2)   // 131328
#define FEXP       (DIM + NTRI)              // 131840, divisible by 32
#define BATCH_SZ   256
#define KTILE      32
#define KSPLIT     8
#define KCHUNK     (FEXP / KSPLIT)           // 16480 = 515 * 32
#define MTILE      16
#define NWAVES     4
#define NTILE      (NWAVES * 16)             // 64
#define BLOCK_THR  128
#define A_STRIDE   40                        // bf16 elems: 80B row, 20-bank stride
#define WF_STRIDE  36                        // fp32 elems: 144B row, 36-bank stride

typedef __attribute__((ext_vector_type(16))) __bf16 v16bf;
typedef __attribute__((ext_vector_type(8)))  float  v8f;

union FragBits { uint4 u[2]; v16bf v; };
union PackBF4  { __bf16 h[4]; uint2 u; };

typedef __attribute__((address_space(3))) char lds_char;
__device__ __forceinline__ unsigned lds_off(const void* p) {
    return (unsigned)(unsigned long long)(lds_char*)p;
}

// Async DMA: 16 bytes global -> LDS, tracked by ASYNCcnt (CDNA5).
__device__ __forceinline__ void async_copy_b128(unsigned ldsDst, const float* gsrc) {
    unsigned long long ga = (unsigned long long)(const void*)gsrc;
    asm volatile("global_load_async_to_lds_b128 %0, %1, off"
                 :: "v"(ldsDst), "v"(ga) : "memory");
}
__device__ __forceinline__ void wait_async0() {
    asm volatile("s_wait_asynccnt 0x0" ::: "memory");
}

// off(i) = number of triu pairs in rows < i  (row i holds j = i..DIM-1)
__device__ __forceinline__ int triu_off(int i) {
    return (i * (2 * DIM + 1 - i)) >> 1;
}

// Precompute triu_indices(DIM): t -> (i, j). Shared by both layers.
__global__ void build_triu_indices(int2* __restrict__ tab) {
    int t = blockIdx.x * blockDim.x + threadIdx.x;
    if (t >= NTRI) return;
    float disc = (float)((2 * DIM + 1) * (2 * DIM + 1) - 8 * t);
    int i = (int)((((float)(2 * DIM + 1)) - sqrtf(disc)) * 0.5f);
    if (i < 0) i = 0;
    if (i > DIM - 1) i = DIM - 1;
    while (i + 1 <= DIM - 1 && triu_off(i + 1) <= t) ++i;   // exact fixup
    while (i > 0 && triu_off(i) > t) --i;
    int j = i + (t - triu_off(i));
    tab[t] = make_int2(i, j);
}

// Y[m][n] = bias[n] : split-K partials are atomically accumulated on top.
__global__ void bias_init(float* __restrict__ Y, const float* __restrict__ bias) {
    int idx = blockIdx.x * blockDim.x + threadIdx.x;
    if (idx < BATCH_SZ * DIM) Y[idx] = bias[idx & (DIM - 1)];
}

// One block = 16 rows x 64 cols x 1/8th of K; 4 waves, one 16x16 tile each.
__global__ __launch_bounds__(BLOCK_THR)
void quad_layer(const float* __restrict__ X,     // [BATCH_SZ, DIM]
                const float* __restrict__ W,     // [DIM, FEXP] row-major
                float*       __restrict__ Y,     // [BATCH_SZ, DIM] (bias-filled)
                const int2*  __restrict__ tab)   // [NTRI]
{
    __shared__ float  xs[MTILE][DIM];                 // 32 KB: block's x rows
    __shared__ __bf16 aT[2][MTILE][A_STRIDE];         // expanded A tile (bf16)
    __shared__ float  wTf[2][NTILE][WF_STRIDE];       // weight tile (fp32, async)

    const int tid    = threadIdx.x;
    const int mBase  = blockIdx.x * MTILE;
    const int nBase  = blockIdx.y * NTILE;
    const int kBegin = blockIdx.z * KCHUNK;
    const int kEnd   = kBegin + KCHUNK;

    // Cooperative load of 16 contiguous x rows (float4-coalesced).
    {
        const float4* src = (const float4*)(X + (size_t)mBase * DIM);
        float4*       dst = (float4*)&xs[0][0];
        for (int idx = tid; idx < MTILE * DIM / 4; idx += BLOCK_THR)
            dst[idx] = src[idx];
    }
    __syncthreads();

    const int lane = tid & 31;
    const int wv   = tid >> 5;
    const int mRow = lane & 15;     // A row per ISA layout
    const int half = lane >> 4;     // K-half (A/B) and M-half (C/D) select

    v8f acc = {0.f, 0.f, 0.f, 0.f, 0.f, 0.f, 0.f, 0.f};

    // Per-thread staging coordinates.
    const int aM = tid >> 3;              // 0..15 : A-tile row
    const int aK = (tid & 7) * 4;         // 0..28 : 4 consecutive k per thread
    const int wR = tid >> 1;              // 0..63 : weight row within N tile
    const int wH = (tid & 1) * 16;        // row half (64B halves of 128B rows)
    const float* wSrc   = W + (size_t)(nBase + wR) * FEXP + wH;
    const float* xrow   = &xs[aM][0];
    const unsigned ldsW0 = lds_off(&wTf[0][wR][wH]);
    const unsigned ldsW1 = lds_off(&wTf[1][wR][wH]);

    // Stage tile at absolute k0 into LDS buffer `buf`: A expansion + async W.
    auto stage = [&](int k0, int buf) {
        PackBF4 p;
        if (k0 < DIM) {                    // uniform branch; only kseg 0, 16 iters
            p.h[0] = (__bf16)xrow[k0 + aK + 0];
            p.h[1] = (__bf16)xrow[k0 + aK + 1];
            p.h[2] = (__bf16)xrow[k0 + aK + 2];
            p.h[3] = (__bf16)xrow[k0 + aK + 3];
        } else {
            const int2* tp = tab + (k0 - DIM) + aK;          // 4 pairs = 32B
            int4 ta = *(const int4*)tp;                      // (i0,j0,i1,j1)
            int4 tb = *(const int4*)(tp + 2);                // (i2,j2,i3,j3)
            p.h[0] = (__bf16)(xrow[ta.x] * xrow[ta.y]);
            p.h[1] = (__bf16)(xrow[ta.z] * xrow[ta.w]);
            p.h[2] = (__bf16)(xrow[tb.x] * xrow[tb.y]);
            p.h[3] = (__bf16)(xrow[tb.z] * xrow[tb.w]);
            __builtin_prefetch(tp + 2 * KTILE, 0, 1);        // next table chunk
        }
        *(uint2*)&aT[buf][aM][aK] = p.u;                     // one b64 store

        // W tile: async DMA global -> LDS (fp32, 64B per thread).
        const unsigned dstBase = buf ? ldsW1 : ldsW0;
        const float*   src     = wSrc + k0;
        #pragma unroll
        for (int q = 0; q < 4; ++q)
            async_copy_b128(dstBase + 16 * q, src + 4 * q);
        if (k0 + KTILE < kEnd)                               // global_prefetch_b8
            __builtin_prefetch(src + KTILE, 0, 1);
    };

    stage(kBegin, 0);                      // prologue fills buffer 0

    int bufCur = 0;
    for (int k0 = kBegin; k0 < kEnd; k0 += KTILE, bufCur ^= 1) {
        wait_async0();                     // this wave's W rows have landed
        __syncthreads();                   // publish A tile across waves

        if (k0 + KTILE < kEnd)             // overlap next DMA with this WMMA
            stage(k0 + KTILE, bufCur ^ 1);

        // --- Gather fragments, contiguous k = 16*half + s in A and B ------------------------
        FragBits a;
        // A 16x32 bf16: VGPR0-3 = K half*16..+7, VGPR4-7 = K half*16+8..+15
        a.u[0] = *(const uint4*)&aT[bufCur][mRow][half * 16];
        a.u[1] = *(const uint4*)&aT[bufCur][mRow][half * 16 + 8];

        // B 32x16 bf16: lane n holds K = half*16..+15, ascending 2 per VGPR.
        const int col = wv * 16 + (lane & 15);
        const float4* wp = (const float4*)&wTf[bufCur][col][half * 16];
        float4 w0 = wp[0], w1 = wp[1], w2 = wp[2], w3 = wp[3];
        v16bf bv;
        bv[0]  = (__bf16)w0.x; bv[1]  = (__bf16)w0.y;
        bv[2]  = (__bf16)w0.z; bv[3]  = (__bf16)w0.w;
        bv[4]  = (__bf16)w1.x; bv[5]  = (__bf16)w1.y;
        bv[6]  = (__bf16)w1.z; bv[7]  = (__bf16)w1.w;
        bv[8]  = (__bf16)w2.x; bv[9]  = (__bf16)w2.y;
        bv[10] = (__bf16)w2.z; bv[11] = (__bf16)w2.w;
        bv[12] = (__bf16)w3.x; bv[13] = (__bf16)w3.y;
        bv[14] = (__bf16)w3.z; bv[15] = (__bf16)w3.w;

        acc = __builtin_amdgcn_wmma_f32_16x16x32_bf16(
                  /*neg_a=*/false, a.v, /*neg_b=*/false, bv,
                  /*c_mod=*/(short)0, acc, /*reuse_a=*/false, /*reuse_b=*/false);
    }

    // --- Epilogue: accumulate split-K partial into Y (native f32 atomics) ---
    const int nGlob = nBase + wv * 16 + (lane & 15);
    #pragma unroll
    for (int r = 0; r < 8; ++r) {
        int m = mBase + r + 8 * half;
        __hip_atomic_fetch_add(&Y[(size_t)m * DIM + nGlob], acc[r],
                               __ATOMIC_RELAXED, __HIP_MEMORY_SCOPE_AGENT);
    }
}

extern "C" void kernel_launch(void* const* d_in, const int* in_sizes, int n_in,
                              void* d_out, int out_size, void* d_ws, size_t ws_size,
                              hipStream_t stream) {
    (void)in_sizes; (void)n_in; (void)out_size; (void)ws_size;

    const float* x  = (const float*)d_in[0];
    const float* W0 = (const float*)d_in[1];
    const float* b0 = (const float*)d_in[2];
    const float* W1 = (const float*)d_in[3];
    const float* b1 = (const float*)d_in[4];
    float* out = (float*)d_out;

    // Workspace: [triu pair table | h buffer]
    int2* tab = (int2*)d_ws;
    size_t tabBytes = (((size_t)NTRI * sizeof(int2)) + 255) & ~(size_t)255;
    float* h = (float*)((char*)d_ws + tabBytes);     // [BATCH_SZ, DIM]

    build_triu_indices<<<(NTRI + 255) / 256, 256, 0, stream>>>(tab);

    dim3 grid(BATCH_SZ / MTILE, DIM / NTILE, KSPLIT);    // (16, 8, 8)

    bias_init<<<(BATCH_SZ * DIM + 255) / 256, 256, 0, stream>>>(h, b0);
    quad_layer<<<grid, BLOCK_THR, 0, stream>>>(x, W0, h, tab);

    bias_init<<<(BATCH_SZ * DIM + 255) / 256, 256, 0, stream>>>(out, b1);
    quad_layer<<<grid, BLOCK_THR, 0, stream>>>(h, W1, out, tab);
}
